// LocalAggregator_10316511445322
// MI455X (gfx1250) — matrix operand order repaired and checked by reading, not compiled, not run on hardware
//
#include <hip/hip_runtime.h>

typedef __attribute__((ext_vector_type(16))) _Float16 v16h;
typedef __attribute__((ext_vector_type(8)))  float    v8f;

#define GRID_RES   0.5f
#define PCM_X     -50.0f
#define PCM_Y     -50.0f
#define PCM_Z      -4.0f
#define SCALE_MULT 3.0f
#define R_MIN      1
#define R_MAX      18
#define LOG2E      1.4426950408889634f

// ---------------------------------------------------------------------------
// Prep 1: per-gaussian packed params (64B struct), padded to nchunk*32 entries:
//  [0..2] mean xyz, [3] opacity,
//  [4..9] quadratic-form coefficients with (-0.5 * log2e) pre-folded:
//     p2 = power*log2e = h00*dx^2 + h11*dy^2 + h22*dz^2
//                      + h01*dx*dy + h02*dx*dz + h12*dy*dz
//  [10] radius (int), [11..13] mean voxel int xyz
// Padding entries (m >= M): radius = -1 so the Chebyshev test always fails
// (cheb >= 0 > -1)  ->  alpha == 0, no contribution anywhere.
// ---------------------------------------------------------------------------
__global__ void la_prep_gauss(const float* __restrict__ means,
                              const float* __restrict__ opas,
                              const float* __restrict__ scales,
                              const float* __restrict__ cov,
                              float* __restrict__ gp, int M, int Mpad) {
  int m = blockIdx.x * blockDim.x + threadIdx.x;
  if (m >= Mpad) return;
  float* o  = gp + (size_t)m * 16;
  int*   oi = (int*)o;
  if (m >= M) {
    for (int t = 0; t < 16; ++t) o[t] = 0.0f;
    oi[10] = -1;               // sentinel: mask always false
    return;
  }
  const float* c = cov + (size_t)m * 9;
  float a00 = c[0], a01 = c[1], a02 = c[2];
  float a11 = c[4], a12 = c[5], a22 = c[8];
  float co00 = a11 * a22 - a12 * a12;
  float co01 = a02 * a12 - a01 * a22;
  float co02 = a01 * a12 - a02 * a11;
  float det  = a00 * co00 + a01 * co01 + a02 * co02;
  float rdet = 1.0f / det;
  float i00 = co00 * rdet;
  float i01 = co01 * rdet;
  float i02 = co02 * rdet;
  float i11 = (a00 * a22 - a02 * a02) * rdet;
  float i12 = (a02 * a01 - a00 * a12) * rdet;
  float i22 = (a00 * a11 - a01 * a01) * rdet;

  // fold -0.5 (diag) / -1.0 (cross, absorbs the 2x) and log2(e)
  const float kd = -0.5f * LOG2E;
  const float kc = -1.0f * LOG2E;

  float mx = means[m * 3 + 0];
  float my = means[m * 3 + 1];
  float mz = means[m * 3 + 2];
  float s  = fmaxf(scales[m * 3 + 0], fmaxf(scales[m * 3 + 1], scales[m * 3 + 2]));
  int rad  = (int)ceilf(s * SCALE_MULT / GRID_RES);
  rad = rad < R_MIN ? R_MIN : (rad > R_MAX ? R_MAX : rad);
  int mi0 = (int)floorf((mx - PCM_X) / GRID_RES);
  int mi1 = (int)floorf((my - PCM_Y) / GRID_RES);
  int mi2 = (int)floorf((mz - PCM_Z) / GRID_RES);

  o[0] = mx;        o[1] = my;        o[2] = mz;        o[3] = opas[m];
  o[4] = kd * i00;  o[5] = kc * i01;  o[6] = kc * i02;  o[7] = kd * i11;
  o[8] = kc * i12;  o[9] = kd * i22;
  oi[10] = rad; oi[11] = mi0; oi[12] = mi1; oi[13] = mi2;
  o[14] = 0.0f; o[15] = 0.0f;
}

// ---------------------------------------------------------------------------
// Prep 2: pack semantics into WMMA B fragments (f16, 16x16x32 layout).
// Fragment (chunk c, channel-block b, lane l) = 16 consecutive halfs:
//   element t -> B[k][chan],  k = c*32 + (l>>4)*8 + (t&7) + (t>=8 ? 16 : 0),  chan = b*16 + (l&15)
// Out-of-range k or chan -> 0 (padded rows have alpha==0 anyway).
// ---------------------------------------------------------------------------
__global__ void la_prep_bpack(const float* __restrict__ sem,
                              _Float16* __restrict__ bp,
                              int M, int C, int nchunk) {
  int idx   = blockIdx.x * blockDim.x + threadIdx.x;
  int total = nchunk * 2 * 32;
  if (idx >= total) return;
  int lane = idx & 31;
  int b    = (idx >> 5) & 1;
  int c    = idx >> 6;
  int chan = b * 16 + (lane & 15);
  int koff = (lane >> 4) * 8;
  _Float16* o = bp + (size_t)idx * 16;
  for (int t = 0; t < 16; ++t) {
    int g = c * 32 + koff + (t & 7) + ((t & 8) << 1);
    float v = (g < M && chan < C) ? sem[(size_t)g * C + chan] : 0.0f;
    o[t] = (_Float16)v;
  }
}

// ---------------------------------------------------------------------------
// Main: one wave = 16 points. Loop over padded gaussians in chunks of 32.
// Gaussian structs stream via immediate-offset loads off one base pointer
// (advances 2KB/chunk). The cheap Chebyshev+quadratic test runs for every
// pair; ALL transcendental work (exp2, log2, reductions) lives inside the
// ~1%-taken mask branch. v_wmma_f32_16x16x32_f16 accumulates logits.
// ---------------------------------------------------------------------------
__global__ void la_main(const float* __restrict__ pts,
                        const float* __restrict__ gp,
                        const _Float16* __restrict__ bp,
                        float* __restrict__ out_logits,
                        float* __restrict__ out_bin,
                        float* __restrict__ out_den,
                        int N, int C, int nchunk) {
  int lid  = threadIdx.x & 31;
  int wave = threadIdx.x >> 5;
  int r    = lid & 15;       // row (for A/alpha) or channel (for B/D)
  int half = lid >> 4;       // which K-half of the A row this lane owns

  int rowbase = (blockIdx.x * (blockDim.x >> 5) + wave) * 16;
  int p  = rowbase + r;
  int pc = p < N ? p : (N - 1);
  float px = pts[(size_t)pc * 3 + 0];
  float py = pts[(size_t)pc * 3 + 1];
  float pz = pts[(size_t)pc * 3 + 2];
  int pi0 = (int)floorf((px - PCM_X) / GRID_RES);
  int pi1 = (int)floorf((py - PCM_Y) / GRID_RES);
  int pi2 = (int)floorf((pz - PCM_Z) / GRID_RES);

  v8f acc0 = {};   // channels 0..15
  v8f acc1 = {};   // channels 16..31
  float density   = 0.0f;
  float log2empty = 0.0f;

  // this lane's base within the chunk: K = half*8 (+16 for the upper block)
  const float*    qbase = gp + (size_t)half * 8 * 16;
  const _Float16* bbase = bp + (size_t)lid * 16;

  for (int c = 0; c < nchunk; ++c) {
    v16h A;
#pragma unroll
    for (int e = 0; e < 16; ++e) {
      // ISA 16-bit A layout: lane half h owns K = h*8 + {0..7} and 16+h*8+{0..7}
      // qbase folds in h*8; constant per-e offset: (e&7) + (e>=8 ? 16 : 0)
      const float* q  = qbase + ((e & 7) + ((e & 8) << 1)) * 16;
      const int*   qi = (const int*)q;
      float dx = px - q[0], dy = py - q[1], dz = pz - q[2];
      // p2 = power * log2e  (coefficients pre-scaled in prep)
      float p2 = q[4] * dx * dx + q[7] * dy * dy + q[9] * dz * dz
               + q[5] * dx * dy + q[6] * dx * dz + q[8] * dy * dz;
      int a0 = pi0 - qi[11]; a0 = a0 < 0 ? -a0 : a0;
      int a1 = pi1 - qi[12]; a1 = a1 < 0 ? -a1 : a1;
      int a2 = pi2 - qi[13]; a2 = a2 < 0 ? -a2 : a2;
      int ch = a0 > a1 ? a0 : a1; ch = ch > a2 ? ch : a2;
      float alpha = 0.0f;
      // padded rows: rad=-1 -> always false
      if ((ch <= qi[10]) && (p2 <= 0.0f)) {
        alpha = q[3] * __builtin_amdgcn_exp2f(p2);
        density += alpha;
        float aclip = fminf(alpha, 1.0f - 1e-6f);
        // v_log_f32 computes log2; masked-out pairs would add log2(1)==0,
        // so skipping them entirely is exact
        log2empty += __builtin_amdgcn_logf(1.0f - aclip);
      }
      A[e] = (_Float16)alpha;
    }

    v16h B0 = *(const v16h*)(bbase);             // channel block 0
    v16h B1 = *(const v16h*)(bbase + 32 * 16);   // channel block 1

    acc0 = __builtin_amdgcn_wmma_f32_16x16x32_f16(
        false, A, false, B0, (short)0, acc0, false, false);
    acc1 = __builtin_amdgcn_wmma_f32_16x16x32_f16(
        false, A, false, B1, (short)0, acc1, false, false);

    qbase += 32 * 16;       // next 32 gaussians (2KB)
    bbase += 2 * 32 * 16;   // next chunk's fragment pair
  }

  // combine the two K-halves of each row for the scalar reductions
  density   += __shfl_xor(density, 16, 32);
  log2empty += __shfl_xor(log2empty, 16, 32);
  if (half == 0 && p < N) {
    out_den[p] = density;
    // prob_empty = 2^(sum of log2 terms)
    out_bin[p] = 1.0f - __builtin_amdgcn_exp2f(log2empty);
  }

  // D layout: lane = channel (r), VGPR i = row half*8+i
#pragma unroll
  for (int i = 0; i < 8; ++i) {
    int row = half * 8 + i;
    int pr  = rowbase + row;
    if (pr < N) {
      if (r < C)      out_logits[(size_t)pr * C + r]      = acc0[i];
      if (16 + r < C) out_logits[(size_t)pr * C + 16 + r] = acc1[i];
    }
  }
}

// ---------------------------------------------------------------------------
extern "C" void kernel_launch(void* const* d_in, const int* in_sizes, int n_in,
                              void* d_out, int out_size, void* d_ws, size_t ws_size,
                              hipStream_t stream) {
  const float* pts    = (const float*)d_in[0];
  const float* means  = (const float*)d_in[1];
  const float* opas   = (const float*)d_in[2];
  const float* sem    = (const float*)d_in[3];
  const float* scales = (const float*)d_in[4];
  const float* cov    = (const float*)d_in[5];

  int N = in_sizes[0] / 3;
  int M = in_sizes[2];
  int C = in_sizes[3] / M;
  int nchunk = (M + 31) / 32;
  int Mpad   = nchunk * 32;

  float*    gp  = (float*)d_ws;                                      // Mpad*16 f32
  _Float16* bpk = (_Float16*)((char*)d_ws + (size_t)Mpad * 16 * 4);  // nchunk*2*32*16 f16

  float* out_logits = (float*)d_out;
  float* out_bin    = out_logits + (size_t)N * C;
  float* out_den    = out_bin + N;

  la_prep_gauss<<<(Mpad + 255) / 256, 256, 0, stream>>>(means, opas, scales, cov,
                                                        gp, M, Mpad);
  int nfrag = nchunk * 2 * 32;
  la_prep_bpack<<<(nfrag + 255) / 256, 256, 0, stream>>>(sem, bpk, M, C, nchunk);

  int waves  = (N + 15) / 16;
  int wpb    = 2;                      // 2 waves / block -> 469 blocks
  int blocks = (waves + wpb - 1) / wpb;
  la_main<<<blocks, wpb * 32, 0, stream>>>(pts, gp, bpk,
                                           out_logits, out_bin, out_den,
                                           N, C, nchunk);
}
